// GINNet_12996571038302
// MI455X (gfx1250) — matrix operand order repaired and checked by reading, not compiled, not run on hardware
//
#include <hip/hip_runtime.h>
#include <hip/hip_bf16.h>
#include <math.h>

// Problem constants from the reference
#define NN 100000     // nodes
#define NE 1600000    // edges
#define NG 64         // graphs
#define BNEPS 1e-5f
#define MT_TOT (NN / 16)   // 6250 row tiles (exact)

typedef __attribute__((ext_vector_type(16))) __bf16 bfx16;  // WMMA A/B operand (8 VGPRs)
typedef __attribute__((ext_vector_type(8)))  float  fx8;    // WMMA C/D operand (8 VGPRs)
typedef __attribute__((ext_vector_type(4)))  unsigned int u32x4;
typedef __attribute__((ext_vector_type(8)))  int i32x8;
typedef __attribute__((ext_vector_type(4)))  int i32x4;

// ---------------------------------------------------------------------------
// Weight conversion: f32 [dout,din] -> bf16 [dout,kpad] (K zero-padded)
// ---------------------------------------------------------------------------
__global__ void k_convert_pad(const float* __restrict__ W, int dout, int din, int kpad,
                              __bf16* __restrict__ dst) {
  int i = blockIdx.x * blockDim.x + threadIdx.x;
  if (i >= dout * kpad) return;
  int o = i / kpad, k = i % kpad;
  float v = (k < din) ? W[o * din + k] : 0.0f;
  dst[i] = (__bf16)v;
}

// ---------------------------------------------------------------------------
// Edge scatter: agg[dst] += x[src]   (f32 atomics; L2-resident targets)
// ---------------------------------------------------------------------------
__global__ void k_scatter3(const float* __restrict__ x, const int* __restrict__ src,
                           const int* __restrict__ dst, float* __restrict__ agg) {
  int e = blockIdx.x * blockDim.x + threadIdx.x;
  if (e >= NE) return;
  int s = src[e], t = dst[e];
  atomicAdd(&agg[t * 3 + 0], x[s * 3 + 0]);
  atomicAdd(&agg[t * 3 + 1], x[s * 3 + 1]);
  atomicAdd(&agg[t * 3 + 2], x[s * 3 + 2]);
}

__global__ void k_scatter4(const float* __restrict__ x, const int* __restrict__ src,
                           const int* __restrict__ dst, int din, float* __restrict__ agg) {
  int d4 = din >> 2;
  long long i = (long long)blockIdx.x * blockDim.x + threadIdx.x;
  if (i >= (long long)NE * d4) return;
  int e = (int)(i / d4);
  int c = ((int)(i % d4)) << 2;
  int s = src[e], t = dst[e];
  const float4 v = *(const float4*)(x + (size_t)s * din + c);
  float* ap = agg + (size_t)t * din + c;
  atomicAdd(ap + 0, v.x); atomicAdd(ap + 1, v.y);
  atomicAdd(ap + 2, v.z); atomicAdd(ap + 3, v.w);
}

// ---------------------------------------------------------------------------
// GIN combine: t_bf16[r,k] = (1+eps)*x[r,k] + agg[r,k]   (K zero-padded)
// ---------------------------------------------------------------------------
__global__ void k_combine(const float* __restrict__ x, const float* __restrict__ agg,
                          const float* __restrict__ epsp, __bf16* __restrict__ t,
                          int din, int kpad) {
  long long i = (long long)blockIdx.x * blockDim.x + threadIdx.x;
  if (i >= (long long)NN * kpad) return;
  int r = (int)(i / kpad), k = (int)(i % kpad);
  float v = 0.0f;
  if (k < din) v = (1.0f + *epsp) * x[(size_t)r * din + k] + agg[(size_t)r * din + k];
  t[i] = (__bf16)v;
}

// ---------------------------------------------------------------------------
// WMMA GEMM v2: Z[NN,dout] = A[NN,K](bf16) * W[dout,K](bf16)^T + bias
//  - 256-thread block = 8 waves; wave = one 16-row tile, NT 16-col tiles.
//  - Weight chunk [NT*16, K] staged into LDS once per block via the Tensor
//    Data Mover (tensor_load_to_lds + s_wait_tensorcnt), 6-arg builtin form.
//  - A fragment reused across NT WMMAs; B fragments come from LDS.
//  - Fused bias + per-column sum/sum-of-squares for BatchNorm.
// ---------------------------------------------------------------------------
template <int NT>
__global__ __launch_bounds__(256)
void k_gemm_bf16(const __bf16* __restrict__ A,   // [NN, K]
                 const __bf16* __restrict__ W,   // [dout, K] row-major
                 const float*  __restrict__ bias,
                 float* __restrict__ Z,          // [NN, dout]
                 int dout, int K,
                 float* __restrict__ colsum, float* __restrict__ colsumsq) {
  __shared__ __bf16 sW[64 * 448];                 // max chunk: 64 cols x K=448 (57 KB)
  const int tid  = threadIdx.x;
  const int wave = tid >> 5;
  const int lane = tid & 31;
  const int r16  = lane & 15;
  const int half = lane >> 4;
  const int colbase = blockIdx.y * (NT * 16);
  const int ncols = NT * 16;

  // ---- stage W[colbase .. colbase+ncols) x [0,K) -> LDS (contiguous block) ----
#if __has_builtin(__builtin_amdgcn_tensor_load_to_lds)
  if (wave == 0) {
    unsigned long long ga = (unsigned long long)(const void*)(W + (size_t)colbase * K);
    unsigned lds = (unsigned)(unsigned long long)(void*)sW;  // generic addr low 32b = LDS offset
    unsigned kk = (unsigned)K, rr = (unsigned)ncols;
    // D# group0: count=1 | lds_addr | global_addr[56:0] | type=2 (ISA 8.3)
    u32x4 g0;
    g0.x = 1u;
    g0.y = lds;
    g0.z = (unsigned)(ga & 0xFFFFFFFFu);
    g0.w = ((unsigned)((ga >> 32) & 0x01FFFFFFu)) | (2u << 30);
    // D# group1: data_size=2B | tensor_dim0=K | tensor_dim1=ncols |
    //            tile_dim0=K | tile_dim1=ncols | tensor_dim0_stride=K (ISA 8.4)
    i32x8 g1;
    g1.s0 = (int)(1u << 16);                                    // data_size code 1 (2B)
    g1.s1 = (int)((kk & 0xFFFFu) << 16);                        // tensor_dim0[15:0]
    g1.s2 = (int)((kk >> 16) | ((rr & 0xFFFFu) << 16));         // dim0[31:16] | dim1[15:0]
    g1.s3 = (int)((rr >> 16) | ((kk & 0xFFFFu) << 16));         // dim1[31:16] | tile_dim0
    g1.s4 = (int)(rr & 0xFFFFu);                                // tile_dim1 | tile_dim2=0
    g1.s5 = (int)kk;                                            // dim0_stride[31:0]
    g1.s6 = 0;
    g1.s7 = 0;
    i32x4 gz4 = {0, 0, 0, 0};                                   // groups 2/3 unused (2D)
    i32x8 gz8 = {0, 0, 0, 0, 0, 0, 0, 0};                       // extended slot (unused)
    __builtin_amdgcn_tensor_load_to_lds(g0, g1, gz4, gz4, gz8, 0);
    __builtin_amdgcn_s_wait_tensorcnt(0);
  }
#else
  {
    const __bf16* gw = W + (size_t)colbase * K;
    for (int idx = tid * 8; idx < ncols * K; idx += 256 * 8)
      *(uint4*)(sW + idx) = *(const uint4*)(gw + idx);
  }
#endif
  __syncthreads();

  const int mtile = blockIdx.x * 8 + wave;
  if (mtile < MT_TOT) {
    const __bf16* arow = A + (size_t)(mtile * 16 + r16) * K;
    const fx8 fz = {0.f, 0.f, 0.f, 0.f, 0.f, 0.f, 0.f, 0.f};
    fx8 acc[NT];
#pragma unroll
    for (int t = 0; t < NT; ++t) acc[t] = fz;

    for (int k0 = 0; k0 < K; k0 += 32) {
      // A frag: lane holds K = k0 + half*16 + {0..15}
      union { uint4 q[2]; bfx16 v; } ua;
      ua.q[0] = *(const uint4*)(arow + k0 + half * 16);
      ua.q[1] = *(const uint4*)(arow + k0 + half * 16 + 8);
      if (k0 + 32 < K) __builtin_prefetch(arow + k0 + 32, 0, 3);   // global_prefetch_b8
#pragma unroll
      for (int t = 0; t < NT; ++t) {
        // B frag from LDS: lane holds column colbase+t*16+r16, K = k0+half*16+0..15
        const __bf16* bp = sW + (size_t)(t * 16 + r16) * K + k0 + half * 16;
        union { uint4 q[2]; bfx16 v; } ub;
        ub.q[0] = *(const uint4*)bp;
        ub.q[1] = *(const uint4*)(bp + 8);
        acc[t] = __builtin_amdgcn_wmma_f32_16x16x32_bf16(false, ua.v, false, ub.v,
                                                         (short)0, acc[t], false, false);
      }
    }

#pragma unroll
    for (int t = 0; t < NT; ++t) {
      const int col = colbase + t * 16 + r16;
      const float bv = bias[col];
      float ps = 0.0f, pq = 0.0f;
#pragma unroll
      for (int v = 0; v < 8; ++v) {          // C layout: VGPR v -> row v + half*8
        int row = mtile * 16 + v + half * 8;
        float z = acc[t][v] + bv;
        Z[(size_t)row * dout + col] = z;
        ps += z; pq += z * z;
      }
      ps += __shfl_xor(ps, 16, 32);          // lanes L and L+16 share a column
      pq += __shfl_xor(pq, 16, 32);
      if (half == 0) {
        atomicAdd(&colsum[col], ps);
        atomicAdd(&colsumsq[col], pq);
      }
    }
  }
}

// ---------------------------------------------------------------------------
// BatchNorm finalize: scale = g*rsqrt(var+eps), shift = be - mean*scale
// ---------------------------------------------------------------------------
__global__ void k_finalize(const float* __restrict__ s, const float* __restrict__ sq,
                           const float* __restrict__ g, const float* __restrict__ be,
                           int d, float* __restrict__ scale, float* __restrict__ shift) {
  int c = blockIdx.x * blockDim.x + threadIdx.x;
  if (c >= d) return;
  float mean = s[c] * (1.0f / NN);
  float var  = sq[c] * (1.0f / NN) - mean * mean;
  float sc = g[c] * rsqrtf(fmaxf(var, 0.0f) + BNEPS);
  scale[c] = sc;
  shift[c] = be[c] - mean * sc;
}

// ---------------------------------------------------------------------------
// BN apply + ReLU; writes f32 (optional) and/or bf16 slice (optional)
// ---------------------------------------------------------------------------
__global__ void k_apply(const float* __restrict__ Z, int d,
                        const float* __restrict__ scale, const float* __restrict__ shift,
                        float* __restrict__ outf,
                        __bf16* __restrict__ outb, int ldb, int coloff) {
  long long i = (long long)blockIdx.x * blockDim.x + threadIdx.x;
  if (i >= (long long)NN * d) return;
  int r = (int)(i / d), c = (int)(i % d);
  float h = fmaxf(Z[i] * scale[c] + shift[c], 0.0f);
  if (outf) outf[i] = h;
  if (outb) outb[(size_t)r * ldb + coloff + c] = (__bf16)h;
}

// ---------------------------------------------------------------------------
// Pooling
// ---------------------------------------------------------------------------
__global__ void k_count(const int* __restrict__ batch, float* __restrict__ cnt) {
  int i = blockIdx.x * blockDim.x + threadIdx.x;
  if (i < NN) atomicAdd(&cnt[batch[i]], 1.0f);
}

__global__ void k_poolsum(const float* __restrict__ h, const int* __restrict__ batch,
                          float* __restrict__ pooled) {
  long long i = (long long)blockIdx.x * blockDim.x + threadIdx.x;
  if (i >= (long long)NN * 384) return;
  int r = (int)(i / 384), c = (int)(i % 384);
  atomicAdd(&pooled[(size_t)batch[r] * 384 + c], h[i]);
}

// ---------------------------------------------------------------------------
// Head: mean pool -> seq2 (384->256, ReLU) -> lin (256->1) -> sigmoid
// ---------------------------------------------------------------------------
__global__ __launch_bounds__(256)
void k_head(const float* __restrict__ pooled, const float* __restrict__ cnt,
            const float* __restrict__ W2, const float* __restrict__ b2,
            const float* __restrict__ Wl, const float* __restrict__ bl,
            float* __restrict__ out) {
  __shared__ float pm[384];
  __shared__ float h2[256];
  __shared__ float red[256];
  int g = blockIdx.x, t = threadIdx.x;
  float inv = 1.0f / fmaxf(cnt[g], 1.0f);
  for (int c = t; c < 384; c += 256) pm[c] = pooled[(size_t)g * 384 + c] * inv;
  __syncthreads();
  float acc = b2[t];
  for (int k = 0; k < 384; ++k) acc = fmaf(W2[(size_t)t * 384 + k], pm[k], acc);
  h2[t] = fmaxf(acc, 0.0f);
  __syncthreads();
  red[t] = Wl[t] * h2[t];
  __syncthreads();
  for (int s = 128; s > 0; s >>= 1) {
    if (t < s) red[t] += red[t + s];
    __syncthreads();
  }
  if (t == 0) out[g] = 1.0f / (1.0f + expf(-(red[0] + bl[0])));
}

// ---------------------------------------------------------------------------
// Launcher
// ---------------------------------------------------------------------------
extern "C" void kernel_launch(void* const* d_in, const int* in_sizes, int n_in,
                              void* d_out, int out_size, void* d_ws, size_t ws_size,
                              hipStream_t stream) {
  (void)out_size; (void)ws_size;
  // ---- resolve flattening order (insertion order vs JAX sorted-key order) ----
  int ix, ie, ib, p0;
  if (in_sizes[0] == 3 * NN) { ix = 0; ie = 1; ib = 2; p0 = 3; }            // x,edge,batch,params
  else                       { ib = 0; ie = 1; p0 = 2; ix = n_in - 1; }     // batch,edge,params,x
  const float* x    = (const float*)d_in[ix];
  const int* edge   = (const int*)d_in[ie];
  const int* srcI   = edge;
  const int* dstI   = edge + NE;
  const int* batch  = (const int*)d_in[ib];

  const bool conv_sorted = (in_sizes[p0 + 1] != 32);
  const float *cW1[6], *cb1[6], *cg1[6], *cbe1[6], *cW2[6], *cb2[6], *cg2[6], *cbe2[6], *ceps[6];
  for (int i = 0; i < 6; ++i) {
    int q = p0 + 9 * i;
    if (!conv_sorted) {  // W1,b1,g1,be1,W2,b2,g2,be2,eps
      cW1[i]=(const float*)d_in[q+0]; cb1[i]=(const float*)d_in[q+1];
      cg1[i]=(const float*)d_in[q+2]; cbe1[i]=(const float*)d_in[q+3];
      cW2[i]=(const float*)d_in[q+4]; cb2[i]=(const float*)d_in[q+5];
      cg2[i]=(const float*)d_in[q+6]; cbe2[i]=(const float*)d_in[q+7];
      ceps[i]=(const float*)d_in[q+8];
    } else {             // W1,W2,b1,b2,be1,be2,eps,g1,g2
      cW1[i]=(const float*)d_in[q+0]; cW2[i]=(const float*)d_in[q+1];
      cb1[i]=(const float*)d_in[q+2]; cb2[i]=(const float*)d_in[q+3];
      cbe1[i]=(const float*)d_in[q+4]; cbe2[i]=(const float*)d_in[q+5];
      ceps[i]=(const float*)d_in[q+6];
      cg1[i]=(const float*)d_in[q+7]; cg2[i]=(const float*)d_in[q+8];
    }
  }
  int t0 = p0 + 54;
  const float *s1W,*s1b,*s1g,*s1be,*s2W,*s2b,*lW,*lb;
  if (in_sizes[t0] == 256) {  // sorted tail: lin(W,b), seq1(W,b,be,g), seq2(W,b)
    lW =(const float*)d_in[t0+0]; lb =(const float*)d_in[t0+1];
    s1W=(const float*)d_in[t0+2]; s1b=(const float*)d_in[t0+3];
    s1be=(const float*)d_in[t0+4]; s1g=(const float*)d_in[t0+5];
    s2W=(const float*)d_in[t0+6]; s2b=(const float*)d_in[t0+7];
  } else {                    // insertion tail: seq1(W,b,g,be), seq2(W,b), lin(W,b)
    s1W=(const float*)d_in[t0+0]; s1b=(const float*)d_in[t0+1];
    s1g=(const float*)d_in[t0+2]; s1be=(const float*)d_in[t0+3];
    s2W=(const float*)d_in[t0+4]; s2b=(const float*)d_in[t0+5];
    lW =(const float*)d_in[t0+6]; lb =(const float*)d_in[t0+7];
  }

  // ---- workspace carve (all 256B aligned) ----
  char* wp = (char*)d_ws;
  auto carve = [&](size_t bytes) -> void* {
    void* p = (void*)wp;
    wp += (bytes + 255) & ~(size_t)255;
    return p;
  };
  float*  hA   = (float*)carve((size_t)NN * 128 * 4);
  float*  hB   = (float*)carve((size_t)NN * 128 * 4);
  float*  agg  = (float*)carve((size_t)NN * 128 * 4);
  __bf16* tbf  = (__bf16*)carve((size_t)NN * 128 * 2);
  float*  zb   = (float*)carve((size_t)NN * 384 * 4);
  __bf16* catb = (__bf16*)carve((size_t)NN * 448 * 2);
  float*  stats = (float*)carve(1024 * 4);   // colsum[512] | colsumsq[512]
  float*  scale = (float*)carve(512 * 4);
  float*  shift = (float*)carve(512 * 4);
  float*  pooled = (float*)carve((size_t)(NG * 384 + NG) * 4);
  float*  cnt = pooled + NG * 384;

  static const int dins[6]  = {3, 32, 32, 64, 64, 128};
  static const int dhids[6] = {32, 32, 64, 64, 128, 128};
  __bf16 *wb1[6], *wb2[6];
  for (int i = 0; i < 6; ++i) {
    int kp1 = dins[i] < 32 ? 32 : dins[i];
    wb1[i] = (__bf16*)carve((size_t)dhids[i] * kp1 * 2);
    wb2[i] = (__bf16*)carve((size_t)dhids[i] * dhids[i] * 2);
  }
  __bf16* wbs1 = (__bf16*)carve((size_t)384 * 448 * 2);

  // ---- convert weights to bf16 ----
  for (int i = 0; i < 6; ++i) {
    int kp1 = dins[i] < 32 ? 32 : dins[i];
    int n1 = dhids[i] * kp1;
    k_convert_pad<<<(n1 + 255) / 256, 256, 0, stream>>>(cW1[i], dhids[i], dins[i], kp1, wb1[i]);
    int n2 = dhids[i] * dhids[i];
    k_convert_pad<<<(n2 + 255) / 256, 256, 0, stream>>>(cW2[i], dhids[i], dhids[i], dhids[i], wb2[i]);
  }
  k_convert_pad<<<(384 * 448 + 255) / 256, 256, 0, stream>>>(s1W, 384, 448, 448, wbs1);

  auto launch_gemm = [&](const __bf16* Aq, const __bf16* Wq, const float* bq,
                         float* Zq, int dout, int K) {
    dim3 grd((MT_TOT + 7) / 8, (unsigned)((dout + 63) / 64));
    if (dout == 32)
      k_gemm_bf16<2><<<grd, 256, 0, stream>>>(Aq, Wq, bq, Zq, dout, K, stats, stats + 512);
    else
      k_gemm_bf16<4><<<grd, 256, 0, stream>>>(Aq, Wq, bq, Zq, dout, K, stats, stats + 512);
  };

  // ---- 6 GIN conv layers ----
  const float* hx = x;
  float* hnext = hA;
  int catoff = 0;
  for (int i = 0; i < 6; ++i) {
    const int din = dins[i], dh = dhids[i];
    const int kp1 = din < 32 ? 32 : din;
    (void)hipMemsetAsync(agg, 0, (size_t)NN * din * 4, stream);
    if (din == 3) {
      k_scatter3<<<(NE + 255) / 256, 256, 0, stream>>>(hx, srcI, dstI, agg);
    } else {
      long long tot = (long long)NE * (din / 4);
      k_scatter4<<<(unsigned)((tot + 255) / 256), 256, 0, stream>>>(hx, srcI, dstI, din, agg);
    }
    {
      long long tot = (long long)NN * kp1;
      k_combine<<<(unsigned)((tot + 255) / 256), 256, 0, stream>>>(hx, agg, ceps[i], tbf, din, kp1);
    }
    // sublayer 1: Linear(W1)+BN+ReLU
    (void)hipMemsetAsync(stats, 0, 1024 * 4, stream);
    launch_gemm(tbf, wb1[i], cb1[i], zb, dh, kp1);
    k_finalize<<<1, dh, 0, stream>>>(stats, stats + 512, cg1[i], cbe1[i], dh, scale, shift);
    {
      long long tot = (long long)NN * dh;
      k_apply<<<(unsigned)((tot + 255) / 256), 256, 0, stream>>>(zb, dh, scale, shift,
                                                                 (float*)nullptr, tbf, dh, 0);
    }
    // sublayer 2: Linear(W2)+BN+ReLU -> f32 h_next + bf16 concat slice
    (void)hipMemsetAsync(stats, 0, 1024 * 4, stream);
    launch_gemm(tbf, wb2[i], cb2[i], zb, dh, dh);
    k_finalize<<<1, dh, 0, stream>>>(stats, stats + 512, cg2[i], cbe2[i], dh, scale, shift);
    {
      long long tot = (long long)NN * dh;
      k_apply<<<(unsigned)((tot + 255) / 256), 256, 0, stream>>>(zb, dh, scale, shift,
                                                                 hnext, catb, 448, catoff);
    }
    catoff += dh;
    hx = hnext;
    hnext = (hnext == hA) ? hB : hA;
  }

  // ---- seq1: 448 -> 384 + BN + ReLU (in-place on zb) ----
  (void)hipMemsetAsync(stats, 0, 1024 * 4, stream);
  launch_gemm(catb, wbs1, s1b, zb, 384, 448);
  k_finalize<<<1, 384, 0, stream>>>(stats, stats + 512, s1g, s1be, 384, scale, shift);
  {
    long long tot = (long long)NN * 384;
    k_apply<<<(unsigned)((tot + 255) / 256), 256, 0, stream>>>(zb, 384, scale, shift,
                                                               zb, (__bf16*)nullptr, 0, 0);
  }

  // ---- global mean pool + head ----
  (void)hipMemsetAsync(pooled, 0, (size_t)(NG * 384 + NG) * 4, stream);
  k_count<<<(NN + 255) / 256, 256, 0, stream>>>(batch, cnt);
  {
    long long tot = (long long)NN * 384;
    k_poolsum<<<(unsigned)((tot + 255) / 256), 256, 0, stream>>>(zb, batch, pooled);
  }
  k_head<<<NG, 256, 0, stream>>>(pooled, cnt, s2W, s2b, lW, lb, (float*)d_out);
}
